// MoELLamaMLP_17093969838308
// MI455X (gfx1250) — compile-verified, matrix operand order checked by reading, not run on hardware
//
#include <hip/hip_runtime.h>
#include <hip/hip_bf16.h>

// ---------------- problem constants (from reference) ----------------
#define D_DIM   1024
#define F_DIM   2752
#define NEXP    8
#define TOKENS  1024           // 2*512
#define TM      32             // tokens per tile
#define FK      32             // F columns per chunk
#define NCHUNK  (F_DIM / FK)   // 86
#define NSLICE  8              // F-slices per (expert,tile) for parallelism
#define CPS     ((NCHUNK + NSLICE - 1) / NSLICE)  // 11 chunks per slice

// LDS strides (ushort elements), all multiples of 8 -> 16B-aligned b128 reads
#define XS        1032   // s_x row stride (D + 8 pad)
#define WS_STRIDE 1032   // s_w col stride for gate/up chunk (K-major, K=1024)
#define HS        40     // s_h row stride (FK + 8 pad)
#define WD_STRIDE 40     // s_w col stride for down chunk (K=32 + 8 pad)

typedef __attribute__((ext_vector_type(16))) __bf16       v16bf;
typedef __attribute__((ext_vector_type(2)))  __bf16       v2bf;
typedef __attribute__((ext_vector_type(8)))  float        v8f;
typedef __attribute__((ext_vector_type(4)))  unsigned int u32x4;
typedef __attribute__((ext_vector_type(2)))  unsigned int u32x2;
typedef __attribute__((ext_vector_type(4)))  float        f32x4;

union BFrag { v16bf v; u32x4 q[2]; };

// pack two fp32 into one dword of bf16 (hardware cvt where available)
__device__ __forceinline__ unsigned bfpk(float a, float b) {
#if __has_builtin(__builtin_amdgcn_cvt_pk_bf16_f32)
  union { v2bf v; unsigned u; } c;
  c.v = __builtin_amdgcn_cvt_pk_bf16_f32(a, b);
  return c.u;
#else
  union { __bf16 h[2]; unsigned u; } c;
  c.h[0] = (__bf16)a; c.h[1] = (__bf16)b;
  return c.u;
#endif
}

// Load one 16x32 (A, rows rc..rc+15) or 32x16 (B, cols rc..rc+15) bf16 fragment
// from LDS; K contiguous with `stride` between rows/cols.
// Per ISA layout: lanes 0-15 hold K = k0+0..7 (q0) and k0+16..23 (q1);
// lanes 16-31 hold K = k0+8..15 and k0+24..31.
__device__ __forceinline__ v16bf load_frag(const unsigned short* s, int rc,
                                           int k0, int stride, int lane) {
  const unsigned short* p = s + (rc + (lane & 15)) * stride + k0 + ((lane >> 4) << 3);
  BFrag f;
  f.q[0] = *(const u32x4*)(p);
  f.q[1] = *(const u32x4*)(p + 16);
  return f.v;
}

__device__ __forceinline__ v8f wmma_bf16(v16bf a, v16bf b, v8f c) {
  return __builtin_amdgcn_wmma_f32_16x16x32_bf16(false, a, false, b,
                                                 (short)0, c, false, false);
}

// ---------------- kernel 0: zero output + expert counters ----------------
__global__ void moe_zero_kernel(float* __restrict__ out, int* __restrict__ cnt) {
  int i = blockIdx.x * blockDim.x + threadIdx.x;   // 262144 float4 = T*D floats
  f32x4 z = {0.0f, 0.0f, 0.0f, 0.0f};
  ((f32x4*)out)[i] = z;
  if (blockIdx.x == 0 && threadIdx.x < NEXP) cnt[threadIdx.x] = 0;
}

// ---------------- kernel 1: gating (one wave32 per token) ----------------
__global__ __launch_bounds__(256) void moe_gate_kernel(
    const float* __restrict__ x, const float* __restrict__ Ws,
    const float* __restrict__ bs, int* __restrict__ cnt,
    int* __restrict__ tok_idx, float* __restrict__ tok_w) {
  const int wv   = threadIdx.x >> 5;
  const int lane = threadIdx.x & 31;
  const int t    = blockIdx.x * 8 + wv;

  float le[NEXP];
#pragma unroll
  for (int e = 0; e < NEXP; ++e) le[e] = 0.0f;

  for (int d = lane; d < D_DIM; d += 32) {
    float xv = x[(size_t)t * D_DIM + d];
    const float* wr = Ws + (size_t)d * NEXP;   // row of 8 floats
#pragma unroll
    for (int e = 0; e < NEXP; ++e) le[e] += xv * wr[e];
  }
#pragma unroll
  for (int e = 0; e < NEXP; ++e) {
#pragma unroll
    for (int off = 16; off > 0; off >>= 1)
      le[e] += __shfl_down(le[e], off, 32);
  }

  if (lane == 0) {
#pragma unroll
    for (int e = 0; e < NEXP; ++e) le[e] += bs[e];
    // top-2 (first index wins ties, matching jax.lax.top_k)
    int e1 = 0; float l1 = le[0];
#pragma unroll
    for (int e = 1; e < NEXP; ++e) if (le[e] > l1) { l1 = le[e]; e1 = e; }
    int e2 = -1; float l2 = -3.4e38f;
#pragma unroll
    for (int e = 0; e < NEXP; ++e)
      if (e != e1 && le[e] > l2) { l2 = le[e]; e2 = e; }
    // softmax over the two selected logits
    float p1 = 1.0f / (1.0f + __expf(l2 - l1));
    float p2 = 1.0f - p1;
    int s1 = atomicAdd(&cnt[e1], 1);
    tok_idx[e1 * TOKENS + s1] = t;
    tok_w [e1 * TOKENS + s1] = p1;
    int s2 = atomicAdd(&cnt[e2], 1);
    tok_idx[e2 * TOKENS + s2] = t;
    tok_w [e2 * TOKENS + s2] = p2;
  }
}

// stage a [D x 32] fp32 chunk (row stride F_DIM) into LDS col-major bf16,
// packing 4 K-values per column into one b64 store.
__device__ __forceinline__ void stage_gu(const float* __restrict__ W, int f0,
                                         unsigned short* s_w, int tid) {
  for (int idx = tid; idx < (D_DIM / 4) * (FK / 4); idx += 128) {
    int k4 = (idx >> 3) << 2;        // 0..1020 step 4
    int n4 = (idx & 7) << 2;         // 0..28  step 4
    const float* p = W + (size_t)k4 * F_DIM + f0 + n4;
    f32x4 r0 = *(const f32x4*)(p);
    f32x4 r1 = *(const f32x4*)(p + F_DIM);
    f32x4 r2 = *(const f32x4*)(p + 2 * F_DIM);
    f32x4 r3 = *(const f32x4*)(p + 3 * F_DIM);
#pragma unroll
    for (int j = 0; j < 4; ++j) {
      u32x2 w; w.x = bfpk(r0[j], r1[j]); w.y = bfpk(r2[j], r3[j]);
      *(u32x2*)&s_w[(n4 + j) * WS_STRIDE + k4] = w;
    }
  }
}

// ---------------- kernel 2: fused expert SwiGLU MLP (bf16 WMMA) ----------------
__global__ __launch_bounds__(128, 1) void moe_mlp_kernel(
    const float* __restrict__ x,  const float* __restrict__ Wg,
    const float* __restrict__ Wu, const float* __restrict__ Wd,
    const int* __restrict__ cnt,  const int* __restrict__ tok_idx,
    const float* __restrict__ tok_w, float* __restrict__ out) {
  __shared__ unsigned short s_x[TM * XS];          // 66.0 KB  bf16 gathered X tile
  __shared__ unsigned short s_w[WD_STRIDE * D_DIM];// 81.9 KB  reused Wg/Wu/Wd chunk
  __shared__ unsigned short s_h[TM * HS];          //  2.5 KB  bf16 h tile
  __shared__ int   s_tidx[TM];
  __shared__ float s_tw[TM];

  const int tid  = threadIdx.x;
  const int lane = tid & 31;
  const int wv   = tid >> 5;               // 4 waves

  const int b     = blockIdx.x;            // E * 32 * 8 = 2048 blocks
  const int e     = b >> 8;
  const int tile  = (b >> 3) & 31;
  const int slice = b & 7;

  const int c   = cnt[e];
  const int m0g = tile * TM;
  if (m0g >= c) return;                    // uniform early-exit for empty tiles

  if (tid < TM) {
    int g = m0g + tid;
    if (g < c) { s_tidx[tid] = tok_idx[e * TOKENS + g]; s_tw[tid] = tok_w[e * TOKENS + g]; }
    else       { s_tidx[tid] = 0;                       s_tw[tid] = 0.0f; }
  }
  __syncthreads();

  // stage gathered X rows as bf16: 2 coalesced b128 reads -> 1 b128 LDS store
  for (int idx = tid; idx < TM * (D_DIM / 8); idx += 128) {
    int m  = idx >> 7;
    int d8 = (idx & 127) << 3;
    const float* p = x + (size_t)s_tidx[m] * D_DIM + d8;
    f32x4 v0 = *(const f32x4*)(p);
    f32x4 v1 = *(const f32x4*)(p + 4);
    u32x4 w;
    w.x = bfpk(v0.x, v0.y); w.y = bfpk(v0.z, v0.w);
    w.z = bfpk(v1.x, v1.y); w.w = bfpk(v1.z, v1.w);
    *(u32x4*)&s_x[m * XS + d8] = w;
  }

  const v8f zf = {0.f, 0.f, 0.f, 0.f, 0.f, 0.f, 0.f, 0.f};
  v8f accd[2][16];                          // 32 tokens x 256 d-cols per wave
#pragma unroll
  for (int i = 0; i < 2; ++i)
#pragma unroll
    for (int j = 0; j < 16; ++j) accd[i][j] = zf;

  const float* Wge = Wg + (size_t)e * D_DIM * F_DIM;
  const float* Wue = Wu + (size_t)e * D_DIM * F_DIM;
  const float* Wde = Wd + (size_t)e * F_DIM * D_DIM;
  const int mt = wv >> 1, nt = wv & 1;      // (M-tile, N-tile) job for gate/up

  for (int cc = 0; cc < CPS; ++cc) {
    int ci = slice * CPS + cc;
    if (ci >= NCHUNK) break;
    int f0 = ci * FK;

    // prefetch hints for the next chunk's weight streams (global_prefetch_b8)
    if (cc + 1 < CPS && ci + 1 < NCHUNK) {
      int f0n = f0 + FK;
      __builtin_prefetch(Wge + (size_t)(tid << 3) * F_DIM + f0n, 0, 1);
      __builtin_prefetch(Wue + (size_t)(tid << 3) * F_DIM + f0n, 0, 1);
      __builtin_prefetch(Wde + (size_t)(f0n + (tid >> 2)) * D_DIM + ((tid & 3) << 8), 0, 1);
    }

    // ---- stage Wg[:, f0:f0+32] col-major bf16 ----
    __syncthreads();
    stage_gu(Wge, f0, s_w, tid);
    __syncthreads();

    // ---- g = X @ Wg-chunk : 32 WMMAs along D ----
    v8f accg = zf;
#pragma unroll 8
    for (int kt = 0; kt < D_DIM; kt += 32) {
      v16bf a  = load_frag(s_x, mt * 16, kt, XS,        lane);
      v16bf bb = load_frag(s_w, nt * 16, kt, WS_STRIDE, lane);
      accg = wmma_bf16(a, bb, accg);
    }
    __syncthreads();

    // ---- stage Wu chunk (reuse s_w) ----
    stage_gu(Wue, f0, s_w, tid);
    __syncthreads();

    // ---- u = X @ Wu-chunk ----
    v8f accu = zf;
#pragma unroll 8
    for (int kt = 0; kt < D_DIM; kt += 32) {
      v16bf a  = load_frag(s_x, mt * 16, kt, XS,        lane);
      v16bf bb = load_frag(s_w, nt * 16, kt, WS_STRIDE, lane);
      accu = wmma_bf16(a, bb, accu);
    }

    // ---- h = silu(g) * u -> s_h (bf16), using C/D lane layout ----
    {
      int mrow = mt * 16 + ((lane >> 4) << 3);
      int ncol = nt * 16 + (lane & 15);
#pragma unroll
      for (int r = 0; r < 8; r += 2) {
        float g0 = accg[r],     u0 = accu[r];
        float g1 = accg[r + 1], u1 = accu[r + 1];
        float h0 = g0 * (1.0f / (1.0f + __expf(-g0))) * u0;
        float h1 = g1 * (1.0f / (1.0f + __expf(-g1))) * u1;
        s_h[(mrow + r) * HS + ncol]     = (unsigned short)(bfpk(h0, h0) & 0xFFFFu);
        s_h[(mrow + r + 1) * HS + ncol] = (unsigned short)(bfpk(h1, h1) & 0xFFFFu);
      }
    }
    __syncthreads();

    // ---- stage Wd[f0:f0+32, :] col-major-by-d bf16 (reuse s_w) ----
    for (int idx = tid; idx < (FK / 4) * (D_DIM / 4); idx += 128) {
      int fl4 = (idx >> 8) << 2;     // 0..28 step 4
      int d4  = (idx & 255) << 2;    // 0..1020 step 4
      const float* p = Wde + (size_t)(f0 + fl4) * D_DIM + d4;
      f32x4 r0 = *(const f32x4*)(p);
      f32x4 r1 = *(const f32x4*)(p + D_DIM);
      f32x4 r2 = *(const f32x4*)(p + 2 * D_DIM);
      f32x4 r3 = *(const f32x4*)(p + 3 * D_DIM);
#pragma unroll
      for (int j = 0; j < 4; ++j) {
        u32x2 w; w.x = bfpk(r0[j], r1[j]); w.y = bfpk(r2[j], r3[j]);
        *(u32x2*)&s_w[(d4 + j) * WD_STRIDE + fl4] = w;
      }
    }
    __syncthreads();

    // ---- y += h @ Wd-chunk : each wave owns 256 d-cols ----
    v16bf a0 = load_frag(s_h, 0,  0, HS, lane);
    v16bf a1 = load_frag(s_h, 16, 0, HS, lane);
#pragma unroll
    for (int n2 = 0; n2 < 16; ++n2) {
      v16bf bb = load_frag(s_w, wv * 256 + n2 * 16, 0, WD_STRIDE, lane);
      accd[0][n2] = wmma_bf16(a0, bb, accd[0][n2]);
      accd[1][n2] = wmma_bf16(a1, bb, accd[1][n2]);
    }
  }

  // ---- weighted combine into out (fp32 atomics; each token in 2 experts x 8 slices)
#pragma unroll
  for (int m2 = 0; m2 < 2; ++m2) {
#pragma unroll
    for (int n2 = 0; n2 < 16; ++n2) {
      int n = wv * 256 + n2 * 16 + (lane & 15);
#pragma unroll
      for (int r = 0; r < 8; ++r) {
        int mr = m2 * 16 + ((lane >> 4) << 3) + r;
        float val = s_tw[mr] * accd[m2][n2][r];
        atomicAdd(out + (size_t)s_tidx[mr] * D_DIM + n, val);
      }
    }
  }
}

// ---------------- launcher ----------------
extern "C" void kernel_launch(void* const* d_in, const int* in_sizes, int n_in,
                              void* d_out, int out_size, void* d_ws, size_t ws_size,
                              hipStream_t stream) {
  const float* x  = (const float*)d_in[0];
  const float* Ws = (const float*)d_in[1];
  const float* bs = (const float*)d_in[2];
  const float* Wg = (const float*)d_in[3];
  const float* Wu = (const float*)d_in[4];
  const float* Wd = (const float*)d_in[5];
  float* out = (float*)d_out;

  char* ws = (char*)d_ws;                       // ~66 KB used
  int*   cnt     = (int*)ws;                    // 8 counters
  int*   tok_idx = (int*)(ws + 256);            // [E][T]
  float* tok_w   = (float*)(ws + 256 + NEXP * TOKENS * sizeof(int));

  moe_zero_kernel<<<dim3((TOKENS * D_DIM) / (256 * 4)), dim3(256), 0, stream>>>(out, cnt);
  moe_gate_kernel<<<dim3(TOKENS / 8), dim3(256), 0, stream>>>(x, Ws, bs, cnt, tok_idx, tok_w);
  moe_mlp_kernel<<<dim3(NEXP * 32 * NSLICE), dim3(128), 0, stream>>>(
      x, Wg, Wu, Wd, cnt, tok_idx, tok_w, out);
}
